// Hierarchical_Attention_44994077393156
// MI455X (gfx1250) — compile-verified, matrix operand order checked by reading, not run on hardware
//
#include <hip/hip_runtime.h>

// ---------------------------------------------------------------------------
// Types for CDNA5 (gfx1250) wave32 WMMA
// ---------------------------------------------------------------------------
typedef _Float16 f16;
typedef __attribute__((ext_vector_type(16))) _Float16 v16h;
typedef __attribute__((ext_vector_type(8)))  _Float16 v8h;
typedef __attribute__((ext_vector_type(8)))  float    v8f;

// Problem constants
#define BATCH 16
#define NQ    1024
#define NKTOK 384
#define DIM   768
#define RED   256
#define CATD  1536

// ---------------------------------------------------------------------------
// Fragment loader for 16-bit A/B operands of v_wmma_f32_16x16x32_f16.
// CDNA5 layout: lane l holds row (l&15); element j of the v16h maps to
//   K = (j%8) + 8*(l>>4) + 16*(j/8)
// => two contiguous 16-byte (b128) loads per lane.
// ---------------------------------------------------------------------------
__device__ __forceinline__ v16h ld_frag(const f16* __restrict__ rowp, int k0, int hi) {
  union { v16h v; v8h p[2]; } u;
  const v8h* q = (const v8h*)(rowp + k0 + 8 * hi);
  u.p[0] = q[0];   // K = k0 + 8*hi + [0..7]
  u.p[1] = q[2];   // K = k0 + 8*hi + 16 + [0..7]
  return u.v;
}

__device__ __forceinline__ v8f wmma_f16(v16h a, v16h b, v8f c) {
  return __builtin_amdgcn_wmma_f32_16x16x32_f16(false, a, false, b, (short)0, c,
                                                false, false);
}

// Store one 16x16 f32 C/D tile (+optional per-column bias, optional f16 copy).
__device__ __forceinline__ void store_tile(const v8f& acc, int m0, int n0,
                                           int r, int hi,
                                           const float* __restrict__ bias,
                                           float* __restrict__ Cf32, long baseC, int ldc,
                                           f16* __restrict__ Cf16, long baseC16, int ldc16) {
  const int n = n0 + r;
  const float bv = bias ? bias[n] : 0.0f;
#pragma unroll
  for (int v = 0; v < 8; ++v) {
    const int m = m0 + v + 8 * hi;
    const float val = acc[v] + bv;
    if (Cf32) Cf32[baseC + (long)m * ldc + n] = val;
    if (Cf16) Cf16[baseC16 + (long)m * ldc16 + n] = (f16)val;
  }
}

// ---------------------------------------------------------------------------
// Generic batched GEMM:  C[b] = A[b] (MxK, row-major f16)
//                               x B[b]^T (B stored NxK row-major f16)
// 2x2 register blocking: each wave owns a 32x32 output tile (4 WMMA tiles),
// reusing each fragment twice -> 2 b128 loads per WMMA.
// grid = (M/32, N/256, batch), block = 256 (8 waves).
// ---------------------------------------------------------------------------
__global__ void gemm_nt_kernel(const f16* __restrict__ A, long strideA,
                               const f16* __restrict__ Bm, long strideB,
                               const float* __restrict__ bias,
                               float* __restrict__ Cf32, long strideC, int ldc,
                               f16* __restrict__ Cf16, long strideC16, int ldc16,
                               int K) {
  const int b    = blockIdx.z;
  const int lane = threadIdx.x & 31;
  const int wave = threadIdx.x >> 5;
  const int r    = lane & 15;
  const int hi   = lane >> 4;
  const int m0   = blockIdx.x * 32;
  const int n0   = (blockIdx.y * 8 + wave) * 32;

  const f16* Ap0 = A + (long)b * strideA + (long)(m0 + r) * K;
  const f16* Ap1 = Ap0 + 16L * K;
  const f16* Bp0 = Bm + (long)b * strideB + (long)(n0 + r) * K;
  const f16* Bp1 = Bp0 + 16L * K;

  v8f acc00 = {}, acc01 = {}, acc10 = {}, acc11 = {};
  for (int k0 = 0; k0 < K; k0 += 32) {
    __builtin_prefetch(Ap0 + k0 + 256, 0, 3);
    __builtin_prefetch(Ap1 + k0 + 256, 0, 3);
    __builtin_prefetch(Bp0 + k0 + 256, 0, 3);
    __builtin_prefetch(Bp1 + k0 + 256, 0, 3);
    v16h a0 = ld_frag(Ap0, k0, hi);
    v16h a1 = ld_frag(Ap1, k0, hi);
    v16h b0 = ld_frag(Bp0, k0, hi);
    v16h b1 = ld_frag(Bp1, k0, hi);
    acc00 = wmma_f16(a0, b0, acc00);
    acc01 = wmma_f16(a0, b1, acc01);
    acc10 = wmma_f16(a1, b0, acc10);
    acc11 = wmma_f16(a1, b1, acc11);
  }

  const long baseC   = (long)b * strideC;
  const long baseC16 = (long)b * strideC16;
  store_tile(acc00, m0,      n0,      r, hi, bias, Cf32, baseC, ldc, Cf16, baseC16, ldc16);
  store_tile(acc01, m0,      n0 + 16, r, hi, bias, Cf32, baseC, ldc, Cf16, baseC16, ldc16);
  store_tile(acc10, m0 + 16, n0,      r, hi, bias, Cf32, baseC, ldc, Cf16, baseC16, ldc16);
  store_tile(acc11, m0 + 16, n0 + 16, r, hi, bias, Cf32, baseC, ldc, Cf16, baseC16, ldc16);
}

// ---------------------------------------------------------------------------
// Fused scores + row-softmax:  out = softmax(Q @ Kc^T * scale, axis=-1), f16.
// Q: [b, N1, 256] f16 (centered), Kc: [b, N2, 256] f16 (centered).
// Each wave processes PAIRS of adjacent column tiles sharing the A fragment.
// grid = (N1/16, B), block = 256 (8 waves). Dynamic LDS: 16*N2 floats.
// ---------------------------------------------------------------------------
__global__ void score_softmax_kernel(const f16* __restrict__ Q,
                                     const f16* __restrict__ Kc,
                                     f16* __restrict__ out,
                                     int N1, int N2, float scale) {
  extern __shared__ float s[];  // [16][N2]
  const int b    = blockIdx.y;
  const int m0   = blockIdx.x * 16;
  const int lane = threadIdx.x & 31;
  const int wave = threadIdx.x >> 5;
  const int r    = lane & 15;
  const int hi   = lane >> 4;

  const f16* Ap = Q + ((long)b * N1 + m0 + r) * RED;
  const int ntiles = N2 >> 4;  // always even here (64 or 24)

  for (int jt = wave * 2; jt < ntiles; jt += 16) {
    const f16* Bp0 = Kc + ((long)b * N2 + jt * 16 + r) * RED;
    const f16* Bp1 = Bp0 + 16L * RED;
    v8f acc0 = {}, acc1 = {};
#pragma unroll
    for (int k0 = 0; k0 < RED; k0 += 32) {
      v16h a  = ld_frag(Ap, k0, hi);
      v16h b0 = ld_frag(Bp0, k0, hi);
      v16h b1 = ld_frag(Bp1, k0, hi);
      acc0 = wmma_f16(a, b0, acc0);
      acc1 = wmma_f16(a, b1, acc1);
    }
#pragma unroll
    for (int v = 0; v < 8; ++v) {
      s[(v + 8 * hi) * N2 + jt * 16 + r]      = acc0[v] * scale;
      s[(v + 8 * hi) * N2 + jt * 16 + 16 + r] = acc1[v] * scale;
    }
  }
  __syncthreads();

  // Row softmax: 16 rows x 16 threads per row.
  const int row = threadIdx.x >> 4;
  const int c   = threadIdx.x & 15;
  float* sr = s + row * N2;

  float mx = -3.4e38f;
  for (int j = c; j < N2; j += 16) mx = fmaxf(mx, sr[j]);
  for (int off = 8; off; off >>= 1) mx = fmaxf(mx, __shfl_xor(mx, off, 16));

  float sum = 0.0f;
  for (int j = c; j < N2; j += 16) {
    float e = __expf(sr[j] - mx);
    sr[j] = e;
    sum += e;
  }
  for (int off = 8; off; off >>= 1) sum += __shfl_xor(sum, off, 16);
  const float inv = 1.0f / sum;

  f16* orow = out + ((long)b * N1 + m0 + row) * N2;
  for (int j = c; j < N2; j += 16) orow[j] = (f16)(sr[j] * inv);
}

// ---------------------------------------------------------------------------
// Subtract per-(b, feature) mean over the token axis; f32 in -> f16 out.
// grid = B, block = RED (256). Coalesced column walks.
// ---------------------------------------------------------------------------
__global__ void center_kernel(const float* __restrict__ Y, f16* __restrict__ out,
                              int Ntok) {
  const int b = blockIdx.x;
  const int rf = threadIdx.x;
  const float* p = Y + (long)b * Ntok * RED + rf;
  float sum = 0.0f;
  for (int n = 0; n < Ntok; ++n) sum += p[(long)n * RED];
  const float mean = sum / (float)Ntok;
  f16* o = out + (long)b * Ntok * RED + rf;
  for (int n = 0; n < Ntok; ++n) o[(long)n * RED] = (f16)(p[(long)n * RED] - mean);
}

// ---------------------------------------------------------------------------
// hidden[b,n] = softmax_n( aw2[b,n,:] . layer_w + layer_b )
// grid = B, block = NKTOK (384).
// ---------------------------------------------------------------------------
__global__ void hidden_kernel(const f16* __restrict__ aw2,
                              const float* __restrict__ lw,
                              const float* __restrict__ lb,
                              float* __restrict__ hidden) {
  __shared__ float w[NKTOK];
  __shared__ float vals[NKTOK];
  __shared__ float red[2];
  const int b = blockIdx.x;
  const int n = threadIdx.x;
  w[n] = lw[n];
  __syncthreads();

  const f16* row = aw2 + ((long)b * NKTOK + n) * NKTOK;
  float acc = lb[0];
  for (int m = 0; m < NKTOK; ++m) acc += (float)row[m] * w[m];
  vals[n] = acc;
  __syncthreads();
  if (n == 0) {
    float mx = -3.4e38f;
    for (int i = 0; i < NKTOK; ++i) mx = fmaxf(mx, vals[i]);
    red[0] = mx;
  }
  __syncthreads();
  const float e = __expf(acc - red[0]);
  vals[n] = e;
  __syncthreads();
  if (n == 0) {
    float sm = 0.0f;
    for (int i = 0; i < NKTOK; ++i) sm += vals[i];
    red[1] = sm;
  }
  __syncthreads();
  hidden[(long)b * NKTOK + n] = e / red[1];
}

// aw3[b,n,m] += hidden[b,m]   (in-place on f16 buffer)
__global__ void add_hidden_kernel(f16* __restrict__ aw3,
                                  const float* __restrict__ hidden, long total) {
  const long idx = (long)blockIdx.x * blockDim.x + threadIdx.x;
  if (idx >= total) return;
  const int m = (int)(idx % NKTOK);
  const int b = (int)(idx / ((long)NQ * NKTOK));
  aw3[idx] = (f16)((float)aw3[idx] + hidden[(long)b * NKTOK + m]);
}

// flat f32 -> f16 convert
__global__ void cvt_kernel(const float* __restrict__ in, f16* __restrict__ out,
                           long n) {
  const long idx = (long)blockIdx.x * blockDim.x + threadIdx.x;
  if (idx < n) out[idx] = (f16)in[idx];
}

// in: [b, N, D] f32 -> out: [b, D, N] f16 (feature-major for value GEMMs)
__global__ void transpose_cvt_kernel(const float* __restrict__ in,
                                     f16* __restrict__ out, int N, int D) {
  const long idx = (long)blockIdx.x * blockDim.x + threadIdx.x;
  const long total = (long)N * D;
  if (idx >= total) return;
  const int n = (int)(idx % N);
  const int d = (int)(idx / N);
  const int b = blockIdx.y;
  out[((long)b * D + d) * N + n] = (f16)in[((long)b * N + n) * D + d];
}

// ---------------------------------------------------------------------------
// Host launcher
// ---------------------------------------------------------------------------
extern "C" void kernel_launch(void* const* d_in, const int* in_sizes, int n_in,
                              void* d_out, int out_size, void* d_ws, size_t ws_size,
                              hipStream_t stream) {
  (void)in_sizes; (void)n_in; (void)out_size; (void)ws_size;

  const float* qf  = (const float*)d_in[0];   // [B, NQ, DIM]
  const float* inf = (const float*)d_in[1];   // [B, NKTOK, DIM]
  const float* W[6]   = {(const float*)d_in[2], (const float*)d_in[4],
                         (const float*)d_in[6], (const float*)d_in[8],
                         (const float*)d_in[10], (const float*)d_in[12]};
  const float* Wb[6]  = {(const float*)d_in[3], (const float*)d_in[5],
                         (const float*)d_in[7], (const float*)d_in[9],
                         (const float*)d_in[11], (const float*)d_in[13]};
  const float* lw   = (const float*)d_in[14];  // [1, 384]
  const float* lb   = (const float*)d_in[15];  // [1]
  const float* Wout = (const float*)d_in[16];  // [768, 1536]
  const float* bout = (const float*)d_in[17];  // [768]

  float* out_cat = (float*)d_out;                                 // [B,NQ,1536]
  float* out_y   = out_cat + (long)BATCH * NQ * CATD;             // [B,NQ,768]

  // ---- workspace carve-up (256B aligned) ----
  char* base = (char*)d_ws;
  size_t off = 0;
  auto alloc = [&](size_t bytes) -> char* {
    char* p = base + off;
    off += (bytes + 255) & ~(size_t)255;
    return p;
  };
  f16*   qf16   = (f16*)alloc((size_t)BATCH * NQ * DIM * 2);
  f16*   if16   = (f16*)alloc((size_t)BATCH * NKTOK * DIM * 2);
  f16*   qfT    = (f16*)alloc((size_t)BATCH * DIM * NQ * 2);
  f16*   ifT    = (f16*)alloc((size_t)BATCH * DIM * NKTOK * 2);
  f16*   W16    = (f16*)alloc((size_t)6 * RED * DIM * 2);
  f16*   Wout16 = (f16*)alloc((size_t)DIM * CATD * 2);
  float* Y      = (float*)alloc((size_t)BATCH * NQ * RED * 4);
  f16*   qc[3], *kc[3];
  for (int i = 0; i < 3; ++i) qc[i] = (f16*)alloc((size_t)BATCH * NQ * RED * 2);
  for (int i = 0; i < 3; ++i) kc[i] = (f16*)alloc((size_t)BATCH * NKTOK * RED * 2);
  f16*   aw1    = (f16*)alloc((size_t)BATCH * NQ * NQ * 2);
  f16*   aw2    = (f16*)alloc((size_t)BATCH * NKTOK * NKTOK * 2);
  f16*   aw3    = (f16*)alloc((size_t)BATCH * NQ * NKTOK * 2);
  float* hid    = (float*)alloc((size_t)BATCH * NKTOK * 4);
  f16*   cat16  = (f16*)alloc((size_t)BATCH * NQ * CATD * 2);

  const float scale = 0.0625f;  // 1/sqrt(256)

  // ---- 1. converts / transposes ----
  {
    long n = (long)BATCH * NQ * DIM;
    cvt_kernel<<<dim3((n + 255) / 256), 256, 0, stream>>>(qf, qf16, n);
    n = (long)BATCH * NKTOK * DIM;
    cvt_kernel<<<dim3((n + 255) / 256), 256, 0, stream>>>(inf, if16, n);
    for (int i = 0; i < 6; ++i) {
      n = (long)RED * DIM;
      cvt_kernel<<<dim3((n + 255) / 256), 256, 0, stream>>>(W[i], W16 + (long)i * RED * DIM, n);
    }
    n = (long)DIM * CATD;
    cvt_kernel<<<dim3((n + 255) / 256), 256, 0, stream>>>(Wout, Wout16, n);
    long t = (long)NQ * DIM;
    transpose_cvt_kernel<<<dim3((t + 255) / 256, BATCH), 256, 0, stream>>>(qf, qfT, NQ, DIM);
    t = (long)NKTOK * DIM;
    transpose_cvt_kernel<<<dim3((t + 255) / 256, BATCH), 256, 0, stream>>>(inf, ifT, NKTOK, DIM);
  }

  // ---- 2. six projections + centering ----
  for (int i = 0; i < 3; ++i) {  // q1,q2,q3
    gemm_nt_kernel<<<dim3(NQ / 32, RED / 256, BATCH), 256, 0, stream>>>(
        qf16, (long)NQ * DIM, W16 + (long)i * RED * DIM, 0L, Wb[i],
        Y, (long)NQ * RED, RED, nullptr, 0L, 0, DIM);
    center_kernel<<<dim3(BATCH), RED, 0, stream>>>(Y, qc[i], NQ);
  }
  for (int i = 0; i < 3; ++i) {  // k1,k2,k3
    gemm_nt_kernel<<<dim3(NKTOK / 32, RED / 256, BATCH), 256, 0, stream>>>(
        if16, (long)NKTOK * DIM, W16 + (long)(i + 3) * RED * DIM, 0L, Wb[i + 3],
        Y, (long)NKTOK * RED, RED, nullptr, 0L, 0, DIM);
    center_kernel<<<dim3(BATCH), RED, 0, stream>>>(Y, kc[i], NKTOK);
  }

  // ---- 3. fused score + softmax ----
  score_softmax_kernel<<<dim3(NQ / 16, BATCH), 256, 16 * NQ * 4, stream>>>(
      qc[0], qc[1], aw1, NQ, NQ, scale);
  score_softmax_kernel<<<dim3(NKTOK / 16, BATCH), 256, 16 * NKTOK * 4, stream>>>(
      kc[0], kc[1], aw2, NKTOK, NKTOK, scale);
  score_softmax_kernel<<<dim3(NQ / 16, BATCH), 256, 16 * NKTOK * 4, stream>>>(
      qc[2], kc[2], aw3, NQ, NKTOK, scale);

  // ---- 4. hidden + broadcast add into aw3 ----
  hidden_kernel<<<dim3(BATCH), NKTOK, 0, stream>>>(aw2, lw, lb, hid);
  {
    long total = (long)BATCH * NQ * NKTOK;
    add_hidden_kernel<<<dim3((total + 255) / 256), 256, 0, stream>>>(aw3, hid, total);
  }

  // ---- 5. attention features -> cat (f32 to d_out, f16 copy) ----
  // af1 = aw1 @ query_feature : B' = qfT [768, 1024]
  gemm_nt_kernel<<<dim3(NQ / 32, DIM / 256, BATCH), 256, 0, stream>>>(
      aw1, (long)NQ * NQ, qfT, (long)DIM * NQ, nullptr,
      out_cat, (long)NQ * CATD, CATD, cat16, (long)NQ * CATD, CATD, NQ);
  // af2 = (aw3 + hidden^T) @ Input_feature : B' = ifT [768, 384]
  gemm_nt_kernel<<<dim3(NQ / 32, DIM / 256, BATCH), 256, 0, stream>>>(
      aw3, (long)NQ * NKTOK, ifT, (long)DIM * NKTOK, nullptr,
      out_cat + DIM, (long)NQ * CATD, CATD, cat16 + DIM, (long)NQ * CATD, CATD, NKTOK);

  // ---- 6. out = cat @ Wout^T + bout ----
  gemm_nt_kernel<<<dim3(NQ / 32, DIM / 256, BATCH), 256, 0, stream>>>(
      cat16, (long)NQ * CATD, Wout16, 0L, bout,
      out_y, (long)NQ * DIM, DIM, nullptr, 0L, 0, CATD);
}
